// SimpleConvOnlyNet_7327214207024
// MI455X (gfx1250) — matrix operand order repaired and checked by reading, not compiled, not run on hardware
//
#include <hip/hip_runtime.h>
#include <cstdint>

// ---------------- problem constants ----------------
#define LINEL 15876              // (128-2)*(128-2) flattened conv output
#define KPAD  15904              // LINEL padded up to a multiple of 32 (497 K-tiles)
#define KT    497                // K tiles of 32
#define NT    516                // N tiles of 16 (516*16 == 8256 == TRI exactly)
#define BATCH 64                 // M = 64 -> 4 tiles of 16

typedef __attribute__((ext_vector_type(16))) __bf16 v16bf;
typedef __attribute__((ext_vector_type(8)))  float  v8f;

union FragAB {
    v16bf    v;
    uint32_t u[8];
    uint4    q[2];
};

// One v_perm_b32: dst = {hi[31:16], lo[31:16]} (bf16 by truncation).
// V_PERM selects from byte pool {S0 = bytes 7..4, S1 = bytes 3..0}.
__device__ __forceinline__ uint32_t pack_bf16_pair(float lo, float hi) {
    return __builtin_amdgcn_perm(__float_as_uint(hi), __float_as_uint(lo), 0x07060302u);
}

__device__ __forceinline__ void pack_b_frag(FragAB& fb,
                                            float4 q0, float4 q1,
                                            float4 q2, float4 q3) {
    fb.u[0] = pack_bf16_pair(q0.x, q0.y);
    fb.u[1] = pack_bf16_pair(q0.z, q0.w);
    fb.u[2] = pack_bf16_pair(q1.x, q1.y);
    fb.u[3] = pack_bf16_pair(q1.z, q1.w);
    fb.u[4] = pack_bf16_pair(q2.x, q2.y);
    fb.u[5] = pack_bf16_pair(q2.z, q2.w);
    fb.u[6] = pack_bf16_pair(q3.x, q3.y);
    fb.u[7] = pack_bf16_pair(q3.z, q3.w);
}

// ---------------------------------------------------------------------------
// Kernel 1: fused conv1(3x3,1->4)+bias+relu, conv2(1x1,4->1)+bias, and pack
// the [64, KPAD] activation matrix as bf16 (zero-padded K) for WMMA A loads.
// ---------------------------------------------------------------------------
__global__ void __launch_bounds__(256)
conv_pack_kernel(const float* __restrict__ x,
                 const float* __restrict__ w1,   // [4,1,3,3]
                 const float* __restrict__ b1,   // [4]
                 const float* __restrict__ w2,   // [1,4,1,1]
                 const float* __restrict__ b2,   // [1]
                 unsigned short* __restrict__ A) // [64, KPAD] bf16
{
    int idx = blockIdx.x * 256 + threadIdx.x;
    const int total = BATCH * KPAD;
    if (idx >= total) return;
    int b = idx / KPAD;
    int p = idx - b * KPAD;

    float h = 0.0f;
    if (p < LINEL) {
        int r = p / 126;
        int c = p - r * 126;
        const float* xb = x + ((size_t)b << 14) + r * 128 + c;
        float xv[3][3];
#pragma unroll
        for (int dy = 0; dy < 3; ++dy)
#pragma unroll
            for (int dx = 0; dx < 3; ++dx)
                xv[dy][dx] = xb[dy * 128 + dx];

        float hsum = b2[0];
#pragma unroll
        for (int oc = 0; oc < 4; ++oc) {
            float acc = b1[oc];
#pragma unroll
            for (int dy = 0; dy < 3; ++dy)
#pragma unroll
                for (int dx = 0; dx < 3; ++dx)
                    acc = fmaf(w1[oc * 9 + dy * 3 + dx], xv[dy][dx], acc);
            hsum = fmaf(w2[oc], fmaxf(acc, 0.0f), hsum);
        }
        h = hsum;
    }
    A[idx] = (unsigned short)(__float_as_uint(h) >> 16);
}

// ---------------------------------------------------------------------------
// Kernel 2: WMMA bf16 GEMM  C[64,8256] = A[64,K] * W[8256,K]^T  (K streamed),
// + bias, + symmetric triu scatter into y[64,128,128].
// One block (8 waves) per N-tile; waves split K round-robin, reduce via LDS.
// Each wave reuses its weight fragment across all 4 M-tiles. Tail K-tile is
// hoisted out of the hot loop (wave-uniform) so the loop has no exec masking.
// ---------------------------------------------------------------------------
__global__ void __launch_bounds__(256)
wmma_gemm_scatter_kernel(const unsigned short* __restrict__ A,   // [64, KPAD] bf16
                         const float* __restrict__ W,            // [8256, LINEL] f32
                         const float* __restrict__ bias,         // [8256]
                         float* __restrict__ out)                // [64,128,128]
{
    const int lane = threadIdx.x & 31;
    const int wave = threadIdx.x >> 5;
    const int n_tile = blockIdx.x;
    const int n = lane & 15;
    const int t = n_tile * 16 + n;          // output feature (triu index)
    const int ksel = lane >> 4;             // 0: low K half, 1: high K half

    const float* wrow = W + (size_t)t * LINEL;

    v8f acc[4] = {};                        // 4 M-tiles of 16x16 f32

    // ---- hot loop: full K-tiles only, no per-lane guards ----
    for (int kt = wave; kt < KT - 1; kt += 8) {
        const int kb = (kt << 5) + (ksel << 4);   // this lane's 16 consecutive K
        const float4* wp = (const float4*)(wrow + kb);
        FragAB fb;
        pack_b_frag(fb, wp[0], wp[1], wp[2], wp[3]);
        // prefetch this lane's next K-tile of the weight stream (stride 8 tiles)
        __builtin_prefetch(wrow + kb + 256, 0, 1);

        const int ka = (kt << 5) + (ksel << 4);
#pragma unroll
        for (int mt = 0; mt < 4; ++mt) {
            const unsigned short* ap = A + (size_t)(mt * 16 + n) * KPAD + ka;
            FragAB fa;
            fa.q[0] = *(const uint4*)ap;          // K +0..7  (or +16..23)
            fa.q[1] = *(const uint4*)(ap + 8);    // K +8..15 (or +24..31)
            acc[mt] = __builtin_amdgcn_wmma_f32_16x16x32_bf16(
                false, fa.v, false, fb.v, (short)0, acc[mt], false, false);
        }
    }

    // ---- tail K-tile (kt = 496, K 15872..15903; zero padding past LINEL) ----
    // 496 % 8 == 0 -> owned by wave 0; wave-uniform branch keeps EXEC full.
    if (wave == 0) {
        const int kt = KT - 1;
        const int kb = (kt << 5) + (ksel << 4);
        float f[16];
#pragma unroll
        for (int e = 0; e < 16; ++e)
            f[e] = (kb + e < LINEL) ? wrow[kb + e] : 0.0f;
        FragAB fb;
#pragma unroll
        for (int e = 0; e < 8; ++e)
            fb.u[e] = pack_bf16_pair(f[2 * e], f[2 * e + 1]);

        const int ka = (kt << 5) + (ksel << 4);
#pragma unroll
        for (int mt = 0; mt < 4; ++mt) {
            const unsigned short* ap = A + (size_t)(mt * 16 + n) * KPAD + ka;
            FragAB fa;
            fa.q[0] = *(const uint4*)ap;
            fa.q[1] = *(const uint4*)(ap + 8);
            acc[mt] = __builtin_amdgcn_wmma_f32_16x16x32_bf16(
                false, fa.v, false, fb.v, (short)0, acc[mt], false, false);
        }
    }

    // ---- cross-wave K reduction through LDS ----
    __shared__ __align__(32) float red[7][32][32];
    if (wave > 0) {
#pragma unroll
        for (int mt = 0; mt < 4; ++mt)
            *(v8f*)&red[wave - 1][lane][mt * 8] = acc[mt];
    }
    __syncthreads();

    if (wave == 0) {
#pragma unroll
        for (int w = 0; w < 7; ++w)
#pragma unroll
            for (int mt = 0; mt < 4; ++mt)
                acc[mt] += *(const v8f*)&red[w][lane][mt * 8];

        // ---- epilogue: bias + symmetric triu scatter ----
        const float bv = bias[t];
        // invert t -> (i, j), i<=j, for N=128: off(i) = i*(257-i)/2
        float s = sqrtf(66049.0f - 8.0f * (float)t);
        int i = (int)((257.0f - s) * 0.5f);
        if (i < 0) i = 0;
        if (i > 127) i = 127;
        while (i > 0 && (i * (257 - i)) / 2 > t) --i;
        while (((i + 1) * (256 - i)) / 2 <= t) ++i;
        const int j = i + (t - (i * (257 - i)) / 2);

        const int mhi = (lane >> 4) << 3;   // lanes 16-31 hold M rows +8
#pragma unroll
        for (int mt = 0; mt < 4; ++mt) {
            union { v8f v; float f[8]; } u;
            u.v = acc[mt];
#pragma unroll
            for (int r = 0; r < 8; ++r) {
                const int b = mt * 16 + mhi + r;
                const float val = u.f[r] + bv;
                float* yb = out + ((size_t)b << 14);
                yb[(i << 7) + j] = val;
                yb[(j << 7) + i] = val;
            }
        }
    }
}

// ---------------------------------------------------------------------------
extern "C" void kernel_launch(void* const* d_in, const int* in_sizes, int n_in,
                              void* d_out, int out_size, void* d_ws, size_t ws_size,
                              hipStream_t stream) {
    (void)in_sizes; (void)n_in; (void)out_size; (void)ws_size;
    const float* x   = (const float*)d_in[0];
    const float* w1  = (const float*)d_in[1];
    const float* b1  = (const float*)d_in[2];
    const float* w2  = (const float*)d_in[3];
    const float* b2  = (const float*)d_in[4];
    const float* ow  = (const float*)d_in[5];
    const float* ob  = (const float*)d_in[6];
    float* out = (float*)d_out;

    unsigned short* Abf = (unsigned short*)d_ws;   // 64*KPAD*2 B ~= 1.94 MB

    const int convElems = BATCH * KPAD;
    conv_pack_kernel<<<(convElems + 255) / 256, 256, 0, stream>>>(x, w1, b1, w2, b2, Abf);
    wmma_gemm_scatter_kernel<<<NT, 256, 0, stream>>>(Abf, ow, ob, out);
}
